// AFNO2D_2293512536364
// MI455X (gfx1250) — matrix operand order repaired and checked by reading, not compiled, not run on hardware
//
#include <hip/hip_runtime.h>
#include <hip/hip_bf16.h>

typedef __attribute__((ext_vector_type(16))) _Float16 v16h;
typedef __attribute__((ext_vector_type(8)))  _Float16 v8h;
typedef __attribute__((ext_vector_type(8)))  float    v8f;

#define NB 8
#define BS 96
#define BSP 104               /* f16 row padded: 208B rows, conflict-free frag loads */
#define CDIM 768
#define HDIM 128
#define WDIM 128
#define TILES_TOTAL 4096      /* (4*16384)/16 pixel tiles of 16 */
#define TILES_PER_WG 16
#define CHUNKS (TILES_TOTAL / TILES_PER_WG)   /* 256 */
#define SHRINK 0.01f

__device__ __forceinline__ v16h cat8(v8h a, v8h b) {
    return __builtin_shufflevector(a, b, 0,1,2,3,4,5,6,7,8,9,10,11,12,13,14,15);
}

// D = A(16x32 f16) * B(32x16 f16) + C(16x16 f32), wave32 WMMA
__global__ __launch_bounds__(192) void afno2d_wmma_kernel(
    const float* __restrict__ x,  const float* __restrict__ w1,
    const float* __restrict__ b1, const float* __restrict__ w2,
    const float* __restrict__ b2, float* __restrict__ out)
{
    __shared__ float     Af32[2][16][BS];   // async-filled fp32 staging (dbl-buffered)
    __shared__ float     Bf32[2][16][BS];   // spatially-negated partner tile
    __shared__ _Float16  Ah[16][BSP];       // f16 WMMA operand tiles (converted once)
    __shared__ _Float16  Bh[16][BSP];
    __shared__ _Float16  Us[16][BSP];       // layer-1 outputs, f16 for layer-2 A op
    __shared__ _Float16  Vs[16][BSP];

    const int n     = blockIdx.x & 7;        // channel block
    const int chunk = blockIdx.x >> 3;       // pixel-tile chunk
    const int tid   = threadIdx.x;
    const int wave  = tid >> 5;              // 0..5  -> 16-col output slab
    const int lane  = tid & 31;

    // ---------- per-lane constants ----------
    const int ncol  = lane & 15;
    const int col   = wave * 16 + ncol;          // output column in [0,96)
    const int lob   = (lane < 16) ? 0 : 8;       // B-operand K offset (K=32 frag, table order)
    const int lo    = (lane < 16) ? 0 : 8;       // A-operand K offset
    const int arow  = lane & 15;                 // A-operand M row
    const int mbase = (lane >> 4) * 8;           // C/D M base for this lane

    // cooperative staging assignment: 192 threads = 16 rows x 12 groups of 8 elems
    const int ldr = tid / 12;
    const int ldc = (tid % 12) * 8;

    // ---- issue async global->LDS copies for one tile (4 asyncs per wave) ----
    auto issue_tile = [&](int tile, int buf) {
        const int p0  = tile << 4;
        const int bi  = p0 >> 14;
        const int rem = p0 & 16383;
        const int h   = rem >> 7;
        const int w0  = rem & 127;
        const float* srcA = x + (size_t)(p0 + ldr) * CDIM + n*BS + ldc;
        const int hn = (HDIM - h) & (HDIM - 1);
        const int wn = (WDIM - (w0 + ldr)) & (WDIM - 1);
        const float* srcB = x + ((size_t)bi * 16384 + (size_t)hn * WDIM + wn) * CDIM
                              + n*BS + ldc;
        unsigned la = (unsigned)(unsigned long long)&Af32[buf][ldr][ldc];
        unsigned lb = (unsigned)(unsigned long long)&Bf32[buf][ldr][ldc];
        asm volatile("global_load_async_to_lds_b128 %0, %1, off"
                     :: "v"(la),       "v"(srcA)     : "memory");
        asm volatile("global_load_async_to_lds_b128 %0, %1, off"
                     :: "v"(la + 16u), "v"(srcA + 4) : "memory");
        asm volatile("global_load_async_to_lds_b128 %0, %1, off"
                     :: "v"(lb),       "v"(srcB)     : "memory");
        asm volatile("global_load_async_to_lds_b128 %0, %1, off"
                     :: "v"(lb + 16u), "v"(srcB + 4) : "memory");
    };

    // prologue: start tile 0 transfer, then overlap it with weight prep
    issue_tile(chunk * TILES_PER_WG, 0);

    // ---------- fold weights (incl. the 0.5 factor) + convert, once per WG ----------
    const float* w10 = w1 + (size_t)(0*NB + n) * BS * BS;
    const float* w11 = w1 + (size_t)(1*NB + n) * BS * BS;
    const float* w20 = w2 + (size_t)(0*NB + n) * BS * BS;
    const float* w21 = w2 + (size_t)(1*NB + n) * BS * BS;

    v16h W1p[3], W1m[3], W2p[3], W2m[3];
    #pragma unroll
    for (int c = 0; c < 3; ++c) {
        #pragma unroll
        for (int i = 0; i < 8; ++i) {
            const int k0 = 32*c + lob + 2*i + ((i >= 4) ? 8 : 0);
            const int k1 = k0 + 1;
            float a0 = w10[k0*BS + col], a1 = w11[k0*BS + col];
            float a2 = w10[k1*BS + col], a3 = w11[k1*BS + col];
            W1p[c][2*i]   = (_Float16)(0.5f * (a0 + a1));
            W1p[c][2*i+1] = (_Float16)(0.5f * (a2 + a3));
            W1m[c][2*i]   = (_Float16)(0.5f * (a0 - a1));
            W1m[c][2*i+1] = (_Float16)(0.5f * (a2 - a3));
            float c0 = w20[k0*BS + col], c1 = w21[k0*BS + col];
            float c2 = w20[k1*BS + col], c3 = w21[k1*BS + col];
            W2p[c][2*i]   = (_Float16)(0.5f * (c0 + c1));
            W2p[c][2*i+1] = (_Float16)(0.5f * (c2 + c3));
            W2m[c][2*i]   = (_Float16)(0.5f * (c0 - c1));
            W2m[c][2*i+1] = (_Float16)(0.5f * (c2 - c3));
        }
    }
    const float b1rh = 0.5f * b1[(size_t)(0*NB + n)*BS + col];
    const float b1ih = 0.5f * b1[(size_t)(1*NB + n)*BS + col];
    const float b2rh = 0.5f * b2[(size_t)(0*NB + n)*BS + col];

    for (int t = 0; t < TILES_PER_WG; ++t) {
        const int p0  = (chunk * TILES_PER_WG + t) << 4;
        const int buf = t & 1;

        // start next tile's transfer into the other buffer (last iter: harmless
        // redundant re-load so the wait constant stays uniform)
        const int nxt = (t + 1 < TILES_PER_WG) ? t + 1 : t;
        issue_tile(chunk * TILES_PER_WG + nxt, (t + 1) & 1);

        // oldest 4 asyncs (= this tile) must be complete; newer 4 stay in flight
        asm volatile("s_wait_asynccnt 0x4" ::: "memory");
        __syncthreads();

        // ---- one-pass cooperative f32 -> f16 conversion (shared, not per-wave) ----
        {
            const float4 a0 = *(const float4*)&Af32[buf][ldr][ldc];
            const float4 a1 = *(const float4*)&Af32[buf][ldr][ldc + 4];
            const float4 g0 = *(const float4*)&Bf32[buf][ldr][ldc];
            const float4 g1 = *(const float4*)&Bf32[buf][ldr][ldc + 4];
            v8h ha, hb;
            ha[0]=(_Float16)a0.x; ha[1]=(_Float16)a0.y; ha[2]=(_Float16)a0.z; ha[3]=(_Float16)a0.w;
            ha[4]=(_Float16)a1.x; ha[5]=(_Float16)a1.y; ha[6]=(_Float16)a1.z; ha[7]=(_Float16)a1.w;
            hb[0]=(_Float16)g0.x; hb[1]=(_Float16)g0.y; hb[2]=(_Float16)g0.z; hb[3]=(_Float16)g0.w;
            hb[4]=(_Float16)g1.x; hb[5]=(_Float16)g1.y; hb[6]=(_Float16)g1.z; hb[7]=(_Float16)g1.w;
            *(v8h*)&Ah[ldr][ldc] = ha;
            *(v8h*)&Bh[ldr][ldc] = hb;
        }
        __syncthreads();

        // ---- layer 1 (9 WMMAs): biases pre-loaded into the C operand ----
        v8f accAp, accAm, accBm = {};
        #pragma unroll
        for (int r = 0; r < 8; ++r) { accAp[r] = b1rh; accAm[r] = b1ih; }
        #pragma unroll
        for (int c = 0; c < 3; ++c) {
            v16h fa = cat8(*(const v8h*)&Ah[arow][32*c + lo],
                           *(const v8h*)&Ah[arow][32*c + 16 + lo]);
            v16h fb = cat8(*(const v8h*)&Bh[arow][32*c + lo],
                           *(const v8h*)&Bh[arow][32*c + 16 + lo]);
            accAp = __builtin_amdgcn_wmma_f32_16x16x32_f16(false, fa, false, W1p[c],
                                                           (short)0, accAp, false, false);
            accAm = __builtin_amdgcn_wmma_f32_16x16x32_f16(false, fa, false, W1m[c],
                                                           (short)0, accAm, false, false);
            accBm = __builtin_amdgcn_wmma_f32_16x16x32_f16(false, fb, false, W1m[c],
                                                           (short)0, accBm, false, false);
        }

        // ---- fused relu (bias/scale already inside), emit f16 u/v tiles ----
        #pragma unroll
        for (int r = 0; r < 8; ++r) {
            float u = fmaxf(accAp[r] + accBm[r], 0.0f);
            float v = fmaxf(accAm[r] - accBm[r], 0.0f);
            Us[mbase + r][col] = (_Float16)u;
            Vs[mbase + r][col] = (_Float16)v;
        }
        __syncthreads();

        // ---- layer 2 (6 WMMAs): two independent chains, bias in u-chain C ----
        v8f accYu, accYv = {};
        #pragma unroll
        for (int r = 0; r < 8; ++r) { accYu[r] = b2rh; }
        #pragma unroll
        for (int c = 0; c < 3; ++c) {
            v16h fu = cat8(*(const v8h*)&Us[arow][32*c + lo],
                           *(const v8h*)&Us[arow][32*c + 16 + lo]);
            v16h fv = cat8(*(const v8h*)&Vs[arow][32*c + lo],
                           *(const v8h*)&Vs[arow][32*c + 16 + lo]);
            accYu = __builtin_amdgcn_wmma_f32_16x16x32_f16(false, fu, false, W2p[c],
                                                           (short)0, accYu, false, false);
            accYv = __builtin_amdgcn_wmma_f32_16x16x32_f16(false, fv, false, W2m[c],
                                                           (short)0, accYv, false, false);
        }

        // ---- fused softshrink + exact fp32 residual, store ----
        #pragma unroll
        for (int r = 0; r < 8; ++r) {
            float y = accYu[r] + accYv[r];
            float s = (y > SHRINK) ? (y - SHRINK)
                                   : ((y < -SHRINK) ? (y + SHRINK) : 0.0f);
            const int m = mbase + r;
            out[(size_t)(p0 + m) * CDIM + n*BS + col] = s + Af32[buf][m][col];
        }
        __syncthreads();   // protect LDS buffers before next tile's writes
    }
}

extern "C" void kernel_launch(void* const* d_in, const int* in_sizes, int n_in,
                              void* d_out, int out_size, void* d_ws, size_t ws_size,
                              hipStream_t stream) {
    (void)in_sizes; (void)n_in; (void)out_size; (void)d_ws; (void)ws_size;
    const float* x  = (const float*)d_in[0];
    const float* w1 = (const float*)d_in[1];
    const float* b1 = (const float*)d_in[2];
    const float* w2 = (const float*)d_in[3];
    const float* b2 = (const float*)d_in[4];
    float* out = (float*)d_out;

    dim3 grid(NB * CHUNKS);   // 8 * 256 = 2048 workgroups
    dim3 block(192);          // 6 waves: one 16-col output slab each
    afno2d_wmma_kernel<<<grid, block, 0, stream>>>(x, w1, b1, w2, b2, out);
}
